// Renderer_66821101191666
// MI455X (gfx1250) — matrix operand order repaired and checked by reading, not compiled, not run on hardware
//
#include <hip/hip_runtime.h>
#include <hip/hip_bf16.h>

typedef __attribute__((ext_vector_type(16))) _Float16 v16h;
typedef __attribute__((ext_vector_type(8)))  float    v8f;
typedef __attribute__((ext_vector_type(4)))  unsigned int u32x4;
typedef __attribute__((ext_vector_type(4)))  int      i32x4;
typedef unsigned int u32;

#define HH 384
#define WW 384
#define HWPIX (384*384)

// Compiler-only ordering barrier (per-wave LDS tiles; LDS ops from one wave
// complete in order per ISA 15.5, so no s_barrier is needed).
#define CSYNC() asm volatile("" ::: "memory")

union F16Frag { v16h v; u32x4 q[2]; _Float16 h[16]; };

// ---------------------------------------------------------------------------
// Async LDS->global store path (CDNA5 GLOBAL_STORE_ASYNC_FROM_LDS, ASYNCcnt).
// Builtin signature (from hipcc diagnostic): (int4 AS1*, int4 AS3*, imm, imm).
// Falls back to plain b128 copy if the builtin is unavailable.
// ---------------------------------------------------------------------------
#if __has_builtin(__builtin_amdgcn_global_store_async_from_lds_b128)
#define ASYNC_STORES 1
#else
#define ASYNC_STORES 0
#endif

typedef __attribute__((address_space(1))) i32x4 g_i32x4;
typedef __attribute__((address_space(3))) i32x4 l_i32x4;

__device__ inline void copy16B_lds_to_global(_Float16* gdst, const _Float16* lsrc) {
#if ASYNC_STORES
  __builtin_amdgcn_global_store_async_from_lds_b128(
      (g_i32x4*)(void*)gdst, (l_i32x4*)(const void*)lsrc, 0, 0);
#else
  *(u32x4*)(void*)gdst = *(const u32x4*)(const void*)lsrc;
#endif
}
__device__ inline void lds_fence_for_async() {   // prior ds_stores visible to async engine
#if ASYNC_STORES
  asm volatile("s_wait_dscnt 0" ::: "memory");
#endif
}
__device__ inline void async_store_drain() {     // LDS buffer safe to reuse
#if ASYNC_STORES
  asm volatile("s_wait_asynccnt 0" ::: "memory");
#endif
}

// A-fragment (16x32 f16, MxK) per CDNA5 ISA:
//  lanes 0-15 : row M=lane&15, halves = K[kb..kb+7]   and K[kb+16..kb+23]
//  lanes 16-31: row M=lane&15, halves = K[kb+8..kb+15] and K[kb+24..kb+31]
__device__ inline v16h load_a_frag_lds(const _Float16* tile, int rowStride,
                                       int lane, int kcol) {
  int half = lane >> 4, m = lane & 15;
  const _Float16* row = tile + m * rowStride;
  F16Frag f;
  f.q[0] = *(const u32x4*)(row + kcol + half * 8);
  f.q[1] = *(const u32x4*)(row + kcol + 16 + half * 8);
  return f.v;
}

// B-fragment from fragment-major repacked weights: 32B contiguous per lane.
__device__ inline v16h load_b_frag(const u32* __restrict__ wf, int ks, int t,
                                   int NT, int lane) {
  const u32* p = wf + (((size_t)(ks * NT + t) * 32) + lane) * 8;
  F16Frag f;
  f.q[0] = *(const u32x4*)p;
  f.q[1] = *(const u32x4*)(p + 4);
  return f.v;
}

__device__ inline float sigmoidf_(float x) { return 1.0f / (1.0f + __expf(-x)); }

// ---------------------------------------------------------------------------
// Repack f32 weights -> wave32 WMMA B-fragment layout (f16 pairs in u32).
//  mode==0: src row-major [Kact x Nact]                       (MLP weights)
//  mode==1: src OIHW, logical K = tap*CIN + ci (tap=ky*3+kx)  (conv weights)
// B layout: n = t*16 + (lane&15); VGPR j holds K = ks*32 + (j>=4?16:0) + (lane>=16?8:0) + 2(j&3) (+1)
// ---------------------------------------------------------------------------
__global__ void repack_kernel(const float* __restrict__ src, u32* __restrict__ dst,
                              int KSTEPS, int Kact, int Nact, int NT, int mode, int CIN) {
  int tid = blockIdx.x * 256 + threadIdx.x;
  if (tid >= KSTEPS * NT * 256) return;
  int j  = tid & 7;
  int l  = (tid >> 3) & 31;
  int t  = (tid >> 8) % NT;
  int ks = (tid >> 8) / NT;
  int n  = t * 16 + (l & 15);
  int k0 = ks * 32 + ((j >> 2) * 16) + ((l >> 4) * 8) + (j & 3) * 2;
  float f01[2] = {0.f, 0.f};
  for (int e = 0; e < 2; ++e) {
    int kk = k0 + e;
    if (n < Nact && kk < Kact) {
      if (mode == 0) f01[e] = src[(size_t)kk * Nact + n];
      else {
        int tap = kk / CIN, ci = kk % CIN;   // w[o][ci][ky][kx]
        f01[e] = src[((size_t)n * CIN + ci) * 9 + tap];
      }
    }
  }
  union { _Float16 h[2]; u32 u; } pk;
  pk.h[0] = (_Float16)f01[0]; pk.h[1] = (_Float16)f01[1];
  dst[tid] = pk.u;
}

// ---------------------------------------------------------------------------
// Fused NeRF-style MLP: PE(xyz) -> 63->128 relu -> 128->128 relu ->
// concat(PE(dir)) -> 160->32, mask, write HWC f16 feature map
// (rdmp[sample*32 + n], sample = pixel*4 + p  ==  HWC with ch = p*32+n).
// One 16-sample tile per wave; per-wave LDS tile [16][192].
// ---------------------------------------------------------------------------
__global__ __launch_bounds__(128) void mlp_kernel(
    const float* __restrict__ zbufs, const float* __restrict__ ray,
    const int* __restrict__ isTrain,
    const u32* __restrict__ w1f, const float* __restrict__ b1,
    const u32* __restrict__ w2f, const float* __restrict__ b2,
    const u32* __restrict__ w3f, const float* __restrict__ b3,
    _Float16* __restrict__ rdmp) {
  __shared__ __align__(16) _Float16 Tsh[4][16][192];
  __shared__ float Msh[4][16];
  int wave = threadIdx.x >> 5, lane = threadIdx.x & 31;
  int tile = blockIdx.x * 4 + wave;
  int base = tile * 16;
  int s = lane & 15, hf = lane >> 4;
  int sample = base + s;
  int pixel = sample >> 2, p = sample & 3;
  float ox = ray[pixel*7+0], oy = ray[pixel*7+1], oz = ray[pixel*7+2];
  float dx = ray[pixel*7+3], dy = ray[pixel*7+4], dz = ray[pixel*7+5];
  float cs = ray[pixel*7+6];
  float z  = zbufs[pixel*4+p];
  float tz = z / cs;
  float xyz[3]  = { ox + dx*tz, oy + dy*tz, oz + dz*tz };
  float dvec[3] = { dx, dy, dz };
  if (hf == 0) {
    float thresh = (isTrain[0] != 0) ? 0.2f : 0.0f;
    Msh[wave][s] = (z > thresh) ? 1.0f : 0.0f;
  }
  _Float16* T = &Tsh[wave][0][0];

  // xyz PE: [x,y,z, sin(xd*2^l) d-major (30), cos (30), pad] -> cols 0..63
  for (int j = 0; j < 32; ++j) {
    int f = hf * 32 + j;
    float v;
    if (f < 3)       v = xyz[f];
    else if (f < 33) { int i = f - 3;  v = __sinf(xyz[i/10] * (float)(1 << (i%10))); }
    else if (f < 63) { int i = f - 33; v = __cosf(xyz[i/10] * (float)(1 << (i%10))); }
    else             v = 0.f;
    T[s*192 + f] = (_Float16)v;
  }
  // dir PE: 27 values + pad -> physical cols 160..191 (logical K 128..159)
  for (int j = 0; j < 16; ++j) {
    int f = hf * 16 + j;
    float v;
    if (f < 3)       v = dvec[f];
    else if (f < 15) { int i = f - 3;  v = __sinf(dvec[i/4] * (float)(1 << (i%4))); }
    else if (f < 27) { int i = f - 15; v = __cosf(dvec[i/4] * (float)(1 << (i%4))); }
    else             v = 0.f;
    T[s*192 + 160 + f] = (_Float16)v;
  }
  CSYNC();

  const v8f vzero = {0,0,0,0,0,0,0,0};
  int nl = lane & 15;

  // ---- layer 1: [16x64] @ [64x128], bias+relu, back to cols 0..127
  {
    v16h a0 = load_a_frag_lds(T, 192, lane, 0);
    v16h a1 = load_a_frag_lds(T, 192, lane, 32);
    CSYNC();
    #pragma unroll
    for (int t = 0; t < 8; ++t) {
      v8f c = vzero;
      c = __builtin_amdgcn_wmma_f32_16x16x32_f16(false, a0, false, load_b_frag(w1f,0,t,8,lane), (short)0, c, false, false);
      c = __builtin_amdgcn_wmma_f32_16x16x32_f16(false, a1, false, load_b_frag(w1f,1,t,8,lane), (short)0, c, false, false);
      int n = t*16 + nl;
      float bv = b1[n];
      #pragma unroll
      for (int r = 0; r < 8; ++r) {
        int m = r + hf*8;                     // C layout: VGPR r -> M = r + 8*(lane>=16)
        T[m*192 + n] = (_Float16)fmaxf(c[r] + bv, 0.0f);
      }
    }
  }
  CSYNC();

  // ---- layer 2: [16x128] @ [128x128], bias+relu, back to cols 0..127
  {
    v16h A[4];
    #pragma unroll
    for (int ks = 0; ks < 4; ++ks) A[ks] = load_a_frag_lds(T, 192, lane, ks*32);
    CSYNC();
    #pragma unroll
    for (int t = 0; t < 8; ++t) {
      v8f c = vzero;
      #pragma unroll
      for (int ks = 0; ks < 4; ++ks)
        c = __builtin_amdgcn_wmma_f32_16x16x32_f16(false, A[ks], false, load_b_frag(w2f,ks,t,8,lane), (short)0, c, false, false);
      int n = t*16 + nl;
      float bv = b2[n];
      #pragma unroll
      for (int r = 0; r < 8; ++r) {
        int m = r + hf*8;
        T[m*192 + n] = (_Float16)fmaxf(c[r] + bv, 0.0f);
      }
    }
  }
  CSYNC();

  // ---- layer 3: [16x160] @ [160x32] (+mask), contiguous HWC tile out
  {
    v16h A[5];
    #pragma unroll
    for (int ks = 0; ks < 4; ++ks) A[ks] = load_a_frag_lds(T, 192, lane, ks*32);
    A[4] = load_a_frag_lds(T, 192, lane, 160);  // logical K=128..159 (dir PE)
    CSYNC();
    _Float16* S = T;                            // reuse as packed [16][32] staging
    #pragma unroll
    for (int t = 0; t < 2; ++t) {
      v8f c = vzero;
      #pragma unroll
      for (int ks = 0; ks < 5; ++ks)
        c = __builtin_amdgcn_wmma_f32_16x16x32_f16(false, A[ks], false, load_b_frag(w3f,ks,t,2,lane), (short)0, c, false, false);
      int n = t*16 + nl;
      float bv = b3[n];
      #pragma unroll
      for (int r = 0; r < 8; ++r) {
        int m = r + hf*8;
        S[m*32 + n] = (_Float16)((c[r] + bv) * Msh[wave][m]);
      }
    }
    CSYNC();
    lds_fence_for_async();
    _Float16* gbase = rdmp + (size_t)base * 32;   // 1KB contiguous per tile
    #pragma unroll
    for (int i = 0; i < 2; ++i) {
      int chunk = lane + 32*i;                    // 64 chunks of 16B
      copy16B_lds_to_global(gbase + chunk*8, S + chunk*8);
    }
    async_store_drain();
  }
}

// ---------------------------------------------------------------------------
// Implicit-GEMM 3x3 SAME conv, HWC f16 activations, f32-accum WMMA.
// K order: k = tap*CIN + ci  => every 32-wide K window = 32 consecutive
// channels at one (dy,dx)-shifted pixel => A-fragments are direct contiguous
// 16B global loads (L2-resident), no im2col staging. 2 M-tiles (32 px) per
// wave for B-fragment reuse. Epilogue: LDS-staged, async b128 stores.
// MODE 0: relu -> f16 HWC | MODE 1: sigmoid * mulsrc -> f16 HWC (MPN fuse)
// MODE 2: sigmoid -> f32 HWC image (Cout=3)
// ---------------------------------------------------------------------------
template<int CIN, int NT, int MODE>
__global__ __launch_bounds__(128) void conv_kernel(
    const _Float16* __restrict__ in, const u32* __restrict__ wf,
    const float* __restrict__ bias, const _Float16* __restrict__ mulsrc,
    _Float16* __restrict__ out16, float* __restrict__ out32, int Cout) {
  constexpr int KSTEPS = (9 * CIN) / 32;
  constexpr int CB = CIN / 32;
  constexpr int NC = NT * 16;                    // padded Cout
  __shared__ __align__(16) _Float16 Ssh[4][16 * NC];
  int wave = threadIdx.x >> 5, lane = threadIdx.x & 31;
  int tile = blockIdx.x * 4 + wave;              // 32-pixel row-segment tile
  int y  = tile / (WW/32);
  int x0 = (tile % (WW/32)) * 32;
  int m = lane & 15, half = lane >> 4, nl = lane & 15;
  const v8f vzero = {0,0,0,0,0,0,0,0};
  v8f c[2][NT];
  #pragma unroll
  for (int mt = 0; mt < 2; ++mt)
    #pragma unroll
    for (int t = 0; t < NT; ++t) c[mt][t] = vzero;

  for (int ks = 0; ks < KSTEPS; ++ks) {
    int tap = ks / CB, cb = ks % CB;
    int ky = tap / 3 - 1, kx = tap % 3 - 1;
    int yy = y + ky;
    bool oky = (yy >= 0) && (yy < HH);
    v16h a[2];
    #pragma unroll
    for (int mt = 0; mt < 2; ++mt) {
      int xx = x0 + mt*16 + m + kx;
      F16Frag f;
      if (oky && xx >= 0 && xx < WW) {
        const _Float16* p = in + ((size_t)yy * WW + xx) * CIN + cb * 32;
        f.q[0] = *(const u32x4*)(p + half * 8);
        f.q[1] = *(const u32x4*)(p + 16 + half * 8);
      } else {
        f.q[0] = u32x4{0,0,0,0}; f.q[1] = u32x4{0,0,0,0};
      }
      a[mt] = f.v;
    }
    #pragma unroll
    for (int t = 0; t < NT; ++t) {
      v16h b = load_b_frag(wf, ks, t, NT, lane);
      c[0][t] = __builtin_amdgcn_wmma_f32_16x16x32_f16(false, a[0], false, b, (short)0, c[0][t], false, false);
      c[1][t] = __builtin_amdgcn_wmma_f32_16x16x32_f16(false, a[1], false, b, (short)0, c[1][t], false, false);
    }
  }

  #pragma unroll
  for (int mt = 0; mt < 2; ++mt) {
    int px0 = x0 + mt*16;
    if (MODE == 2) {
      float bv = (nl < Cout) ? bias[nl] : 0.f;
      #pragma unroll
      for (int r = 0; r < 8; ++r) {
        int mr = r + half*8;
        int pix = y*WW + px0 + mr;
        if (nl < Cout) out32[(size_t)pix*3 + nl] = sigmoidf_(c[mt][0][r] + bv);
      }
    } else {
      _Float16* S = Ssh[wave];
      #pragma unroll
      for (int t = 0; t < NT; ++t) {
        int n = t*16 + nl;
        float bv = bias[n];
        #pragma unroll
        for (int r = 0; r < 8; ++r) {
          int mr = r + half*8;
          float v = c[mt][t][r] + bv;
          if (MODE == 0) v = fmaxf(v, 0.0f);
          else {
            int pix = y*WW + px0 + mr;
            v = sigmoidf_(v) * (float)mulsrc[(size_t)pix * NC + n];
          }
          S[mr*NC + n] = (_Float16)v;
        }
      }
      CSYNC();
      lds_fence_for_async();
      _Float16* gbase = out16 + ((size_t)y*WW + px0) * NC;  // contiguous 16px*NC
      #pragma unroll
      for (int i = 0; i < NT; ++i) {
        int chunk = lane + 32*i;                            // NT*32 chunks of 16B
        copy16B_lds_to_global(gbase + chunk*8, S + chunk*8);
      }
      async_store_drain();                                  // before LDS reuse (mt=1)
    }
  }
}

// ---------------------------------------------------------------------------
extern "C" void kernel_launch(void* const* d_in, const int* in_sizes, int n_in,
                              void* d_out, int out_size, void* d_ws, size_t ws_size,
                              hipStream_t stream) {
  (void)in_sizes; (void)n_in; (void)out_size; (void)ws_size;
  const float* zbufs   = (const float*)d_in[0];
  const float* ray     = (const float*)d_in[1];
  const int*   isTrain = (const int*)  d_in[4];
  const float* w1  = (const float*)d_in[6];
  const float* b1  = (const float*)d_in[7];
  const float* w2  = (const float*)d_in[8];
  const float* b2  = (const float*)d_in[9];
  const float* w3  = (const float*)d_in[10];
  const float* b3  = (const float*)d_in[11];
  const float* mw1 = (const float*)d_in[12];
  const float* mb1 = (const float*)d_in[13];
  const float* mw2 = (const float*)d_in[14];
  const float* mb2 = (const float*)d_in[15];
  const float* uw1 = (const float*)d_in[16];
  const float* ub1 = (const float*)d_in[17];
  const float* uw2 = (const float*)d_in[18];
  const float* ub2 = (const float*)d_in[19];
  const float* uw3 = (const float*)d_in[20];
  const float* ub3 = (const float*)d_in[21];

  char* ws = (char*)d_ws;
  size_t off = 0;
  auto alloc = [&](size_t bytes) -> void* {
    void* p = ws + off;
    off = (off + bytes + 255) & ~(size_t)255;
    return p;
  };
  const size_t MAP128 = (size_t)HWPIX * 128 * sizeof(_Float16);  // 37.75 MB (HWC)
  _Float16* rdmp = (_Float16*)alloc(MAP128);
  _Float16* c1f  = (_Float16*)alloc(MAP128);
  _Float16* fuse = (_Float16*)alloc(MAP128);
  _Float16* u1   = rdmp;   // rdmp dead after the fuse conv
  _Float16* u2   = c1f;    // c1 dead after the fuse conv
  u32* w1f = (u32*)alloc((size_t) 2*8*256*4);
  u32* w2f = (u32*)alloc((size_t) 4*8*256*4);
  u32* w3f = (u32*)alloc((size_t) 5*2*256*4);
  u32* m1f = (u32*)alloc((size_t)36*8*256*4);
  u32* m2f = (u32*)alloc((size_t)36*8*256*4);
  u32* u1f = (u32*)alloc((size_t)36*4*256*4);
  u32* u2f = (u32*)alloc((size_t)18*4*256*4);
  u32* u3f = (u32*)alloc((size_t)18*1*256*4);

  // Weight repack (tiny; fragment-major, L2-resident thereafter)
  repack_kernel<<< 2*8, 256, 0, stream>>>(w1,  w1f,  2,   63, 128, 8, 0,   0);
  repack_kernel<<< 4*8, 256, 0, stream>>>(w2,  w2f,  4,  128, 128, 8, 0,   0);
  repack_kernel<<< 5*2, 256, 0, stream>>>(w3,  w3f,  5,  155,  32, 2, 0,   0);
  repack_kernel<<<36*8, 256, 0, stream>>>(mw1, m1f, 36, 1152, 128, 8, 1, 128);
  repack_kernel<<<36*8, 256, 0, stream>>>(mw2, m2f, 36, 1152, 128, 8, 1, 128);
  repack_kernel<<<36*4, 256, 0, stream>>>(uw1, u1f, 36, 1152,  64, 4, 1, 128);
  repack_kernel<<<18*4, 256, 0, stream>>>(uw2, u2f, 18,  576,  64, 4, 1,  64);
  repack_kernel<<<18*1, 256, 0, stream>>>(uw3, u3f, 18,  576,   3, 1, 1,  64);

  // Fused PE + 3-layer MLP + mask -> rdmp (HWC f16, [HW][128])
  mlp_kernel<<<9216, 128, 0, stream>>>(zbufs, ray, isTrain,
                                       w1f, b1, w2f, b2, w3f, b3, rdmp);

  // Convs: 4608 32-px tiles / 4 waves per block
  conv_kernel<128,8,0><<<1152,128,0,stream>>>(rdmp, m1f, mb1, nullptr, c1f,  nullptr, 128);
  conv_kernel<128,8,1><<<1152,128,0,stream>>>(c1f,  m2f, mb2, rdmp,    fuse, nullptr, 128);
  conv_kernel<128,4,0><<<1152,128,0,stream>>>(fuse, u1f, ub1, nullptr, u1,   nullptr,  64);
  conv_kernel< 64,4,0><<<1152,128,0,stream>>>(u1,   u2f, ub2, nullptr, u2,   nullptr,  64);
  conv_kernel< 64,1,2><<<1152,128,0,stream>>>(u2,   u3f, ub3, nullptr, nullptr, (float*)d_out, 3);
}
